// IpaBlock_13640816132144
// MI455X (gfx1250) — matrix operand order repaired and checked by reading, not compiled, not run on hardware
//
#include <hip/hip_runtime.h>
#include <hip/hip_bf16.h>
#include <stdint.h>

#define NNODE 10000
#define NEDGE 160000
#define HID   256
#define NH    8
#define HD    32

typedef __attribute__((ext_vector_type(16))) _Float16 v16h;
typedef __attribute__((ext_vector_type(8)))  float    v8f;

union Frag16 { v16h v; uint32_t u[8]; };

// ---------------------------------------------------------------------------
// float atomic max via int punning (works with -inf init, mixed signs)
__device__ __forceinline__ void atomicMaxF(float* addr, float val) {
    if (val >= 0.0f) atomicMax((int*)addr, __float_as_int(val));
    else             atomicMin((unsigned int*)addr, __float_as_uint(val));
}

// ---------------------------------------------------------------------------
// f32 -> f16 elementwise convert
__global__ void cvt_f32_f16(const float* __restrict__ in, _Float16* __restrict__ out, int n) {
    int i = blockIdx.x * blockDim.x + threadIdx.x;
    if (i < n) out[i] = (_Float16)in[i];
}

// ---------------------------------------------------------------------------
// WMMA GEMM: out[M,Nout] = act( A[M,K](f16) @ W[Nout,K]^T(f16) + bias + res )
// One wave computes a 16x64 output tile (4 accumulators) so the A fragment is
// reused across 4 WMMAs per K-step. act: 0=none, 1=SiLU.
__global__ void gemm_wmma_f16(const _Float16* __restrict__ A,
                              const _Float16* __restrict__ W,
                              const float* __restrict__ bias,
                              const float* __restrict__ res,       // may be null
                              float* __restrict__ outF,
                              _Float16* __restrict__ outH,         // may be null
                              int M, int Nout, int K, int act) {
    const int wavesPerBlk = blockDim.x >> 5;
    const int wave  = blockIdx.x * wavesPerBlk + (threadIdx.x >> 5);
    const int groupsN = Nout >> 6;                     // 64-col groups
    const int totalWaves = (M >> 4) * groupsN;
    if (wave >= totalWaves) return;
    const int tM = wave / groupsN;
    const int tG = wave % groupsN;

    const int lane   = threadIdx.x & 31;
    const int laneLo = lane & 15;
    const int hiA = (lane >> 4) * 8;    // A: upper-half lanes take K+8 within group
    const int hiB = (lane >> 4) * 16;   // unused: B loads use hiA (K+8), same order as A

    const _Float16* Arow = A + (size_t)(tM * 16 + laneLo) * K;
    const _Float16* Wrow0 = W + (size_t)(tG * 64 + laneLo) * K;

    v8f c0 = {}, c1 = {}, c2 = {}, c3 = {};
    for (int k0 = 0; k0 < K; k0 += 32) {
        Frag16 a, b0, b1, b2, b3;
#pragma unroll
        for (int j = 0; j < 8; ++j) {
            const int ka = k0 + ((j & 4) << 2) + hiA + ((j & 3) << 1); // {0..7,16..23}+hiA
            a.u[j] = *(const uint32_t*)(Arow + ka);
            const int kb = k0 + ((j & 4) << 2) + hiA + ((j & 3) << 1); // {0..7,16..23}+hiA
            b0.u[j] = *(const uint32_t*)(Wrow0 + kb);
            b1.u[j] = *(const uint32_t*)(Wrow0 + 16 * K + kb);
            b2.u[j] = *(const uint32_t*)(Wrow0 + 32 * K + kb);
            b3.u[j] = *(const uint32_t*)(Wrow0 + 48 * K + kb);
        }
        c0 = __builtin_amdgcn_wmma_f32_16x16x32_f16(false, a.v, false, b0.v, (short)0, c0, false, false);
        c1 = __builtin_amdgcn_wmma_f32_16x16x32_f16(false, a.v, false, b1.v, (short)0, c1, false, false);
        c2 = __builtin_amdgcn_wmma_f32_16x16x32_f16(false, a.v, false, b2.v, (short)0, c2, false, false);
        c3 = __builtin_amdgcn_wmma_f32_16x16x32_f16(false, a.v, false, b3.v, (short)0, c3, false, false);
    }

    v8f acc[4] = {c0, c1, c2, c3};
#pragma unroll
    for (int sub = 0; sub < 4; ++sub) {
        const int nOut = tG * 64 + sub * 16 + laneLo;
        const float bval = bias[nOut];
#pragma unroll
        for (int r = 0; r < 8; ++r) {
            const int mOut = tM * 16 + r + ((lane >> 4) << 3);
            float vv = acc[sub][r] + bval;
            if (res) vv += res[(size_t)mOut * Nout + nOut];
            if (act == 1) vv = vv / (1.0f + expf(-vv));     // SiLU
            outF[(size_t)mOut * Nout + nOut] = vv;
            if (outH) outH[(size_t)mOut * Nout + nOut] = (_Float16)vv;
        }
    }
}

// ---------------------------------------------------------------------------
// Per-edge logits + geometry terms; stores total[E,8], local/raw deltas, atomic max.
__global__ void edge_logits(const float* __restrict__ q, const float* __restrict__ k,
                            const int* __restrict__ src, const int* __restrict__ dst,
                            const float* __restrict__ x, const float* __restrict__ frames,
                            const float* __restrict__ dist, const float* __restrict__ bpp,
                            const float* __restrict__ msa, const float* __restrict__ chem,
                            const float* __restrict__ rel, const float* __restrict__ chain,
                            const float* __restrict__ Wd, const float* __restrict__ bd,
                            const float* __restrict__ Wb, const float* __restrict__ bb,
                            const float* __restrict__ Wm, const float* __restrict__ bm,
                            const float* __restrict__ Wc, const float* __restrict__ bc,
                            const float* __restrict__ Wr, const float* __restrict__ br,
                            const float* __restrict__ Wch, const float* __restrict__ bch,
                            const float* __restrict__ Wori, const float* __restrict__ bori,
                            float* __restrict__ total, float* __restrict__ dloc,
                            float* __restrict__ draw, float* __restrict__ mmax, int E) {
    int e = blockIdx.x * blockDim.x + threadIdx.x;
    if (e >= E) return;
    const int s = src[e];
    const int t = dst[e];

    const float d0 = x[t * 3 + 0] - x[s * 3 + 0];
    const float d1 = x[t * 3 + 1] - x[s * 3 + 1];
    const float d2 = x[t * 3 + 2] - x[s * 3 + 2];
    const float* F = frames + (size_t)s * 9;
    // local[i] = sum_j F[j][i] * d[j]
    const float l0 = F[0] * d0 + F[3] * d1 + F[6] * d2;
    const float l1 = F[1] * d0 + F[4] * d1 + F[7] * d2;
    const float l2 = F[2] * d0 + F[5] * d1 + F[8] * d2;
    float nrm = sqrtf(l0 * l0 + l1 * l1 + l2 * l2);
    nrm = fmaxf(nrm, 1e-6f);
    const float u0 = l0 / nrm, u1 = l1 / nrm, u2 = l2 / nrm;

    draw[e * 3 + 0] = d0; draw[e * 3 + 1] = d1; draw[e * 3 + 2] = d2;
    dloc[e * 3 + 0] = l0; dloc[e * 3 + 1] = l1; dloc[e * 3 + 2] = l2;

    const float dist2 = dist[e] * dist[e];
    const float fb = bpp[e], fm = msa[e], fc = chem[e], fr = rel[e], fch = chain[e];

    const float4* qs = (const float4*)(q + (size_t)s * HID);
    const float4* kd = (const float4*)(k + (size_t)t * HID);
    const float inv = 0.17677669529663687f;   // 1/sqrt(32)

#pragma unroll
    for (int h = 0; h < NH; ++h) {
        float acc = 0.0f;
#pragma unroll
        for (int j = 0; j < 8; ++j) {
            float4 a = qs[h * 8 + j];
            float4 b = kd[h * 8 + j];
            acc += a.x * b.x + a.y * b.y + a.z * b.z + a.w * b.w;
        }
        float tot = acc * inv
                  - (dist2 * Wd[h] + bd[h])
                  + (fb * Wb[h] + bb[h]) + (fm * Wm[h] + bm[h])
                  + (fc * Wc[h] + bc[h]) + (fr * Wr[h] + br[h])
                  + (fch * Wch[h] + bch[h])
                  + 0.1f * tanhf(u0 * Wori[h * 3 + 0] + u1 * Wori[h * 3 + 1]
                                 + u2 * Wori[h * 3 + 2] + bori[h]);
        total[(size_t)e * NH + h] = tot;
        atomicMaxF(mmax + (size_t)s * NH + h, tot);
    }
}

// ---------------------------------------------------------------------------
// exp(total - max[src]) and atomic segment sum
__global__ void edge_exp(const int* __restrict__ src,
                         const float* __restrict__ mmax,
                         float* __restrict__ total,     // in: logits, out: exp
                         float* __restrict__ ssum, int EH) {
    int t = blockIdx.x * blockDim.x + threadIdx.x;
    if (t >= EH) return;
    const int e = t >> 3;
    const int h = t & 7;
    const int s = src[e];
    const float v = expf(total[t] - mmax[(size_t)s * NH + h]);
    total[t] = v;
    atomicAdd(ssum + (size_t)s * NH + h, v);
}

// ---------------------------------------------------------------------------
// weighted-V scatter + displacement scatter. One thread per (edge, dim).
__global__ void edge_scatter(const int* __restrict__ src, const int* __restrict__ dst,
                             const float* __restrict__ ebuf,   // exp weights numerator [E,8]
                             const float* __restrict__ ssum,   // [N,8]
                             const float* __restrict__ vproj,  // [N,256]
                             const float* __restrict__ dloc, const float* __restrict__ draw,
                             float* __restrict__ hupd,         // [N,256]
                             float* __restrict__ displ, float* __restrict__ dispr,
                             int E) {
    int t = blockIdx.x * blockDim.x + threadIdx.x;
    if (t >= E * HD) return;
    const int e = t >> 5;
    const int d = t & 31;
    const int s = src[e];
    const int dd = dst[e];
    float wsum = 0.0f;
#pragma unroll
    for (int h = 0; h < NH; ++h) {
        const float w = ebuf[(size_t)e * NH + h]
                      / fmaxf(ssum[(size_t)s * NH + h], 1e-9f);
        wsum += w;
        atomicAdd(hupd + (size_t)s * HID + h * HD + d,
                  w * vproj[(size_t)dd * HID + h * HD + d]);
    }
    if (d == 0) {
        const float wm = wsum * 0.125f;     // mean over 8 heads
#pragma unroll
        for (int i = 0; i < 3; ++i) {
            atomicAdd(displ + (size_t)s * 3 + i, dloc[(size_t)e * 3 + i] * wm);
            atomicAdd(dispr + (size_t)s * 3 + i, draw[(size_t)e * 3 + i] * wm);
        }
    }
}

// ---------------------------------------------------------------------------
// Gate layer-2 matvec + tanh + coordinate update. One wave32 per node.
__global__ void node_final(const float* __restrict__ g1, const float* __restrict__ Wg2,
                           const float* __restrict__ bg2, const float* __restrict__ x,
                           const float* __restrict__ frames,
                           const float* __restrict__ displ, const float* __restrict__ dispr,
                           float* __restrict__ xout, int N) {
    const int node = blockIdx.x * (blockDim.x >> 5) + (threadIdx.x >> 5);
    if (node >= N) return;
    const int lane = threadIdx.x & 31;
    const float* row = g1 + (size_t)node * HID;
    float acc = 0.0f;
#pragma unroll
    for (int j = 0; j < 8; ++j) acc += row[lane + 32 * j] * Wg2[lane + 32 * j];
#pragma unroll
    for (int o = 16; o > 0; o >>= 1) acc += __shfl_xor(acc, o, 32);
    if (lane == 0) {
        const float gate = tanhf(acc + bg2[0]);
        const float* F = frames + (size_t)node * 9;
        const float dl0 = displ[node * 3 + 0], dl1 = displ[node * 3 + 1], dl2 = displ[node * 3 + 2];
#pragma unroll
        for (int i = 0; i < 3; ++i) {
            const float dg = 0.5f * (F[i * 3 + 0] * dl0 + F[i * 3 + 1] * dl1 + F[i * 3 + 2] * dl2)
                           + 0.5f * dispr[node * 3 + i];
            xout[node * 3 + i] = x[node * 3 + i] + gate * dg;
        }
    }
}

// ---------------------------------------------------------------------------
extern "C" void kernel_launch(void* const* d_in, const int* in_sizes, int n_in,
                              void* d_out, int out_size, void* d_ws, size_t ws_size,
                              hipStream_t stream) {
    const float* h       = (const float*)d_in[0];
    const float* x       = (const float*)d_in[1];
    const int*   src     = (const int*)  d_in[2];
    const int*   dst     = (const int*)  d_in[3];
    const float* dist    = (const float*)d_in[4];
    const float* bpp     = (const float*)d_in[5];
    const float* msa     = (const float*)d_in[6];
    const float* chem    = (const float*)d_in[7];
    const float* rel     = (const float*)d_in[8];
    const float* chain   = (const float*)d_in[9];
    const float* frames  = (const float*)d_in[10];
    const float* Wq = (const float*)d_in[11]; const float* bq = (const float*)d_in[12];
    const float* Wk = (const float*)d_in[13]; const float* bk = (const float*)d_in[14];
    const float* Wv = (const float*)d_in[15]; const float* bv = (const float*)d_in[16];
    const float* Wo = (const float*)d_in[17]; const float* bo = (const float*)d_in[18];
    const float* Wd = (const float*)d_in[19]; const float* bd = (const float*)d_in[20];
    const float* Wb = (const float*)d_in[21]; const float* bb = (const float*)d_in[22];
    const float* Wm = (const float*)d_in[23]; const float* bm = (const float*)d_in[24];
    const float* Wc = (const float*)d_in[25]; const float* bc = (const float*)d_in[26];
    const float* Wr = (const float*)d_in[27]; const float* br = (const float*)d_in[28];
    const float* Wch= (const float*)d_in[29]; const float* bch= (const float*)d_in[30];
    const float* Wori=(const float*)d_in[31]; const float* bori=(const float*)d_in[32];
    const float* Wg1= (const float*)d_in[33]; const float* bg1= (const float*)d_in[34];
    const float* Wg2= (const float*)d_in[35]; const float* bg2= (const float*)d_in[36];

    float* h_out = (float*)d_out;                       // [N, 256]
    float* x_out = (float*)d_out + (size_t)NNODE * HID; // [N, 3]

    // ---- workspace carve-up (256B aligned) ----
    char* ws = (char*)d_ws;
    size_t off = 0;
    auto carve = [&](size_t bytes) {
        char* p = ws + off;
        off += (bytes + 255) & ~(size_t)255;
        return p;
    };
    _Float16* h16    = (_Float16*)carve((size_t)NNODE * HID * 2);
    _Float16* Wq16   = (_Float16*)carve((size_t)HID * HID * 2);
    _Float16* Wk16   = (_Float16*)carve((size_t)HID * HID * 2);
    _Float16* Wv16   = (_Float16*)carve((size_t)HID * HID * 2);
    _Float16* Wo16   = (_Float16*)carve((size_t)HID * HID * 2);
    _Float16* Wg116  = (_Float16*)carve((size_t)HID * HID * 2);
    float*    qb     = (float*)carve((size_t)NNODE * HID * 4);
    float*    kb     = (float*)carve((size_t)NNODE * HID * 4);
    float*    vb     = (float*)carve((size_t)NNODE * HID * 4);
    float*    total  = (float*)carve((size_t)NEDGE * NH * 4);
    float*    dloc   = (float*)carve((size_t)NEDGE * 3 * 4);
    float*    draw   = (float*)carve((size_t)NEDGE * 3 * 4);
    float*    mmax   = (float*)carve((size_t)NNODE * NH * 4);
    float*    ssum   = (float*)carve((size_t)NNODE * NH * 4);
    float*    hupd   = (float*)carve((size_t)NNODE * HID * 4);
    _Float16* hupd16 = (_Float16*)carve((size_t)NNODE * HID * 2);
    _Float16* hout16 = (_Float16*)carve((size_t)NNODE * HID * 2);
    float*    g1buf  = (float*)carve((size_t)NNODE * HID * 4);
    float*    displ  = (float*)carve((size_t)NNODE * 3 * 4);
    float*    dispr  = (float*)carve((size_t)NNODE * 3 * 4);
    (void)ws_size; (void)n_in; (void)in_sizes; (void)out_size;

    // ---- zero / -inf initialization (graph-capture safe) ----
    hipMemsetAsync(mmax, 0xFF, (size_t)NNODE * NH * 4, stream);   // -NaN bits: valid seed for punned max
    hipMemsetAsync(ssum, 0,    (size_t)NNODE * NH * 4, stream);
    hipMemsetAsync(hupd, 0,    (size_t)NNODE * HID * 4, stream);
    hipMemsetAsync(displ, 0,   (size_t)NNODE * 3 * 4, stream);
    hipMemsetAsync(dispr, 0,   (size_t)NNODE * 3 * 4, stream);

    // ---- f16 conversions ----
    {
        const int nH = NNODE * HID, nW = HID * HID;
        cvt_f32_f16<<<(nH + 255) / 256, 256, 0, stream>>>(h,  h16,  nH);
        cvt_f32_f16<<<(nW + 255) / 256, 256, 0, stream>>>(Wq, Wq16, nW);
        cvt_f32_f16<<<(nW + 255) / 256, 256, 0, stream>>>(Wk, Wk16, nW);
        cvt_f32_f16<<<(nW + 255) / 256, 256, 0, stream>>>(Wv, Wv16, nW);
        cvt_f32_f16<<<(nW + 255) / 256, 256, 0, stream>>>(Wo, Wo16, nW);
        cvt_f32_f16<<<(nW + 255) / 256, 256, 0, stream>>>(Wg1, Wg116, nW);
    }

    // ---- q/k/v projections via WMMA (16x64 tile per wave) ----
    const int waves = (NNODE / 16) * (HID / 64);      // 2500 waves
    const int gemmBlocks = (waves + 7) / 8;           // 8 waves / 256-thread block
    gemm_wmma_f16<<<gemmBlocks, 256, 0, stream>>>(h16, Wq16, bq, nullptr, qb, nullptr, NNODE, HID, HID, 0);
    gemm_wmma_f16<<<gemmBlocks, 256, 0, stream>>>(h16, Wk16, bk, nullptr, kb, nullptr, NNODE, HID, HID, 0);
    gemm_wmma_f16<<<gemmBlocks, 256, 0, stream>>>(h16, Wv16, bv, nullptr, vb, nullptr, NNODE, HID, HID, 0);

    // ---- edge logits + geometry + segment max ----
    edge_logits<<<(NEDGE + 255) / 256, 256, 0, stream>>>(
        qb, kb, src, dst, x, frames, dist, bpp, msa, chem, rel, chain,
        Wd, bd, Wb, bb, Wm, bm, Wc, bc, Wr, br, Wch, bch, Wori, bori,
        total, dloc, draw, mmax, NEDGE);

    // ---- exp + segment sum ----
    edge_exp<<<(NEDGE * NH + 255) / 256, 256, 0, stream>>>(src, mmax, total, ssum, NEDGE * NH);

    // ---- weighted-V scatter + displacements ----
    edge_scatter<<<(NEDGE * HD + 255) / 256, 256, 0, stream>>>(
        src, dst, total, ssum, vb, dloc, draw, hupd, displ, dispr, NEDGE);

    // ---- output projection (residual) via WMMA ----
    cvt_f32_f16<<<(NNODE * HID + 255) / 256, 256, 0, stream>>>(hupd, hupd16, NNODE * HID);
    gemm_wmma_f16<<<gemmBlocks, 256, 0, stream>>>(hupd16, Wo16, bo, h, h_out, hout16, NNODE, HID, HID, 0);

    // ---- gate layer 1 (SiLU) via WMMA ----
    gemm_wmma_f16<<<gemmBlocks, 256, 0, stream>>>(hout16, Wg116, bg1, nullptr, g1buf, nullptr, NNODE, HID, HID, 1);

    // ---- gate layer 2 + tanh + coordinate update ----
    node_final<<<(NNODE + 7) / 8, 256, 0, stream>>>(g1buf, Wg2, bg2, x, frames, displ, dispr, x_out, NNODE);
}
